// WindowMutualSelfAttention_47863115546758
// MI455X (gfx1250) — matrix operand order repaired and checked by reading, not compiled, not run on hardware
//
#include <hip/hip_runtime.h>
#include <math.h>

// ---------------------------------------------------------------------------
// Window mutual/self attention for MI455X (gfx1250, wave32, WMMA bf16)
// ---------------------------------------------------------------------------

#define NTOK   128
#define CDIM   144
#define LDX    160          // K padded (144 -> 160, multiple of 32)
#define NHEADS 6
#define HD     24
#define HDP    32           // head dim padded to 32
#define LDC    320          // concat buffer stride (288 channels, padded)
#define LDBETA 132          // beta (f32) row stride
#define QSCALE 0.2041241452319315f   // 24^-0.5

typedef __attribute__((ext_vector_type(16))) __bf16 v16bf;
typedef __attribute__((ext_vector_type(8)))  __bf16 v8bf;
typedef __attribute__((ext_vector_type(8)))  float  v8f;

// ws layout (bf16 elements)
#define WSELF_OFF 0                      // 576 x 160 (per-head padded qkv_self)
#define WMUT_OFF  (576*160)              // 576 x 160
#define WPROJ_OFF (2*576*160)            // 144 x 288
#define POS_OFF   (2*576*160 + 144*288)  // 128 x 160
#define WS_ELEMS  (2*576*160 + 144*288 + 128*160)   // 246272 elements

static __device__ __forceinline__ v8f zero_acc() {
    v8f z;
#pragma unroll
    for (int i = 0; i < 8; ++i) z[i] = 0.0f;
    return z;
}

// A fragment (16x32 bf16): lanes 0-15 hold K 0..15 contiguous, lanes 16-31
// hold K 16..31.  p already includes (lane>>4)*16.
static __device__ __forceinline__ v16bf load_a_frag(const __bf16* p) {
    return *(const v16bf*)p;
}

// B fragment (32x16 bf16): lanes 0-15 hold K 0..15 contiguous, lanes 16-31
// hold K 16..31.  p already includes (lane>>4)*16.
static __device__ __forceinline__ v16bf load_b_frag(const __bf16* p) {
    return *(const v16bf*)p;
}

// One 16x16 f32 output tile:  D += A(16 x 32*KT) * B(32*KT x 16)
// arow/bcol include the per-lane (lane&15) offset already.
// A row-major stride lda; B stored as rows of "output column" (i.e. B^T
// row-major) with stride ldb, so both sides load contiguously.
template <int KT>
static __device__ __forceinline__ v8f gemm_rowrow(const __bf16* A, int lda, int arow,
                                                  const __bf16* B, int ldb, int bcol,
                                                  int lane, v8f acc) {
    const int ka = (lane >> 4) << 4;
    const int kb = (lane >> 4) << 4;
    const __bf16* ap = A + arow * lda + ka;
    const __bf16* bp = B + bcol * ldb + kb;
#pragma unroll
    for (int kt = 0; kt < KT; ++kt) {
        v16bf af = load_a_frag(ap + kt * 32);
        v16bf bf = load_b_frag(bp + kt * 32);
        acc = __builtin_amdgcn_wmma_f32_16x16x32_bf16(false, af, false, bf,
                                                      (short)0, acc, false, false);
    }
    return acc;
}

// ---------------------------------------------------------------------------
// Prep kernel: bf16-pack weights + positional-encoding table into d_ws.
// qkv weights repacked per (mat, head) with head-dim padded 24->32 and
// K padded 144->160 so every main-kernel B-fragment load is unconditional.
// ---------------------------------------------------------------------------
__global__ void wmsa_prep_kernel(const float* __restrict__ wself,
                                 const float* __restrict__ wmut,
                                 const float* __restrict__ wproj,
                                 __bf16* __restrict__ ws) {
    int i = blockIdx.x * blockDim.x + threadIdx.x;
    const int R0 = 576 * 160;
    const int R2 = 144 * 288;
    const int R3 = 128 * 160;

    if (i < 2 * R0) {
        const float* src = (i < R0) ? wself : wmut;
        int off          = (i < R0) ? WSELF_OFF : WMUT_OFF;
        int ii = (i < R0) ? i : (i - R0);
        int row = ii / 160, c = ii % 160;
        int mh = row >> 5, j = row & 31;        // mh = mat*6 + head
        int mat = mh / 6, h = mh % 6;
        float v = (j < HD && c < CDIM) ? src[(mat * CDIM + h * HD + j) * CDIM + c] : 0.0f;
        ws[off + ii] = (__bf16)v;
        return;
    }
    i -= 2 * R0;
    if (i < R2) {                               // w_proj: (144 x 288) row-major, no pad
        ws[WPROJ_OFF + i] = (__bf16)wproj[i];
        return;
    }
    i -= R2;
    if (i < R3) {                               // sine positional encoding, tiled over 2 halves
        int n = i / 160, c = i % 160;
        float v = 0.0f;
        if (c < CDIM) {
            int n2 = n & 63;
            int hh = n2 >> 3, wwp = n2 & 7;
            const float twopi = 6.283185307179586f;
            float y  = (float)(hh + 1)  / (8.0f + 1e-6f) * twopi;
            float xx = (float)(wwp + 1) / (8.0f + 1e-6f) * twopi;
            int   t2    = (c < 72) ? c : (c - 72);
            float coord = (c < 72) ? y : xx;
            float freq  = powf(10000.0f, (float)(2 * (t2 >> 1)) / 72.0f);
            float ang   = coord / freq;
            v = (t2 & 1) ? cosf(ang) : sinf(ang);
        }
        ws[POS_OFF + i] = (__bf16)v;
    }
}

// ---------------------------------------------------------------------------
// Per-head qkv projection: q/k row-major [128][32] (q pre-scaled), v stored
// transposed [32][128] so alpha@V B-fragments are contiguous.
// 48 output tiles (3 mats x 8 Mtiles x 2 Ntiles), 6 per wave, 5 K-steps each.
// `wave` is an SGPR (readfirstlane) so mat/mtile/ntile stay scalar and the
// store-target selection compiles to s_cbranch instead of EXEC juggling.
// ---------------------------------------------------------------------------
static __device__ __forceinline__ void qkv_stage(const __bf16* src, const __bf16* wq,
                                                 int h, int wave, int lane,
                                                 __bf16* sq, __bf16* sk, __bf16* svT) {
#pragma unroll 1
    for (int i = 0; i < 6; ++i) {
        int t = wave * 6 + i;
        int mat   = __builtin_amdgcn_readfirstlane(t >> 4);
        int rem   = t & 15;
        int mtile = __builtin_amdgcn_readfirstlane(rem >> 1);
        int ntile = __builtin_amdgcn_readfirstlane(rem & 1);
        int arow = mtile * 16 + (lane & 15);
        int brow = (mat * NHEADS + h) * 32 + ntile * 16 + (lane & 15);
        v8f acc = gemm_rowrow<5>(src, LDX, arow, wq, LDX, brow, lane, zero_acc());
        int col   = ntile * 16 + (lane & 15);
        int rbase = mtile * 16 + ((lane >> 4) << 3);
        if (mat == 2) {
#pragma unroll
            for (int r = 0; r < 8; ++r) svT[col * NTOK + rbase + r] = (__bf16)acc[r];
        } else if (mat == 0) {
#pragma unroll
            for (int r = 0; r < 8; ++r) sq[(rbase + r) * HDP + col] = (__bf16)(acc[r] * QSCALE);
        } else {
#pragma unroll
            for (int r = 0; r < 8; ++r) sk[(rbase + r) * HDP + col] = (__bf16)acc[r];
        }
    }
}

// ---------------------------------------------------------------------------
// Main kernel: one block per window.
// ---------------------------------------------------------------------------
__global__ __launch_bounds__(256) void wmsa_main_kernel(
        const float* __restrict__ x, const float* __restrict__ rpb,
        const float* __restrict__ b_proj, const __bf16* __restrict__ ws,
        float* __restrict__ out) {

    __shared__ __bf16 sh_xb [NTOK * LDX];      //  40 KB  x (bf16, K-padded)
    __shared__ __bf16 sh_xpb[NTOK * LDX];      //  40 KB  x + pos
    __shared__ __bf16 sh_cat[NTOK * LDC];      //  80 KB  [mutual(0:144) | self(144:288)]
    __shared__ __bf16 sh_q  [NTOK * HDP];      //   8 KB
    __shared__ __bf16 sh_k  [NTOK * HDP];      //   8 KB
    __shared__ __bf16 sh_vT [HDP * NTOK];      //   8 KB
    __shared__ float  sh_beta[NTOK * LDBETA];  //  66 KB
    __shared__ __bf16 sh_alpha[NTOK * NTOK];   //  32 KB

    const int tid  = threadIdx.x;
    const int lane = tid & 31;
    const int wave = __builtin_amdgcn_readfirstlane(tid >> 5);   // scalar wave id
    const int b    = blockIdx.x;

    const __bf16* wselfp = ws + WSELF_OFF;
    const __bf16* wmutp  = ws + WMUT_OFF;
    const __bf16* wprojp = ws + WPROJ_OFF;
    const __bf16* posp   = ws + POS_OFF;

    // ---- stage 0: stage x and x+pos into LDS as bf16 -----------------------
    for (int i = tid; i < NTOK * LDX; i += 256) {
        int r = i / LDX, c = i % LDX;
        float xv = (c < CDIM) ? x[((size_t)b * NTOK + r) * CDIM + c] : 0.0f;
        sh_xb[i]  = (__bf16)xv;
        sh_xpb[i] = (__bf16)(xv + (float)posp[i]);
    }
    __syncthreads();

    // ======================= self attention (with rpb) ======================
#pragma unroll 1
    for (int h = 0; h < NHEADS; ++h) {
        qkv_stage(sh_xb, wselfp, h, wave, lane, sh_q, sh_k, sh_vT);
        __syncthreads();

        // beta = q @ k^T + rpb  : 64 tiles, 8 per wave, K=32 (1 wmma each)
#pragma unroll 1
        for (int i = 0; i < 8; ++i) {
            int mtile = wave;                       // t = wave*8+i -> t>>3 = wave
            int ntile = i;
            int arow = mtile * 16 + (lane & 15);
            int bcol = ntile * 16 + (lane & 15);
            v8f acc = gemm_rowrow<1>(sh_q, HDP, arow, sh_k, HDP, bcol, lane, zero_acc());
            int m = bcol;
            int dm = m >> 6, hm = (m >> 3) & 7, wm = m & 7;
            int rbase = mtile * 16 + ((lane >> 4) << 3);
#pragma unroll
            for (int r = 0; r < 8; ++r) {
                int n = rbase + r;
                int dn = n >> 6, hn = (n >> 3) & 7, wn = n & 7;
                int idx = (dn - dm + 1) * 225 + (hn - hm + 7) * 15 + (wn - wm + 7);
                sh_beta[n * LDBETA + m] = acc[r] + rpb[idx * NHEADS + h];
            }
        }
        __syncthreads();

        // softmax over 128 cols: wave handles rows wave, wave+8, ...
#pragma unroll 1
        for (int i = 0; i < 16; ++i) {
            int row = wave + (i << 3);
            float v0 = sh_beta[row * LDBETA + lane];
            float v1 = sh_beta[row * LDBETA + lane + 32];
            float v2 = sh_beta[row * LDBETA + lane + 64];
            float v3 = sh_beta[row * LDBETA + lane + 96];
            float mx = fmaxf(fmaxf(v0, v1), fmaxf(v2, v3));
            for (int off = 16; off; off >>= 1) mx = fmaxf(mx, __shfl_xor(mx, off));
            float e0 = __expf(v0 - mx), e1 = __expf(v1 - mx);
            float e2 = __expf(v2 - mx), e3 = __expf(v3 - mx);
            float s = e0 + e1 + e2 + e3;
            for (int off = 16; off; off >>= 1) s += __shfl_xor(s, off);
            float inv = 1.0f / s;
            sh_alpha[row * NTOK + lane]      = (__bf16)(e0 * inv);
            sh_alpha[row * NTOK + lane + 32] = (__bf16)(e1 * inv);
            sh_alpha[row * NTOK + lane + 64] = (__bf16)(e2 * inv);
            sh_alpha[row * NTOK + lane + 96] = (__bf16)(e3 * inv);
        }
        __syncthreads();

        // out_h = alpha @ v : 16 tiles, 2 per wave, K=128 (4 wmma each)
#pragma unroll 1
        for (int i = 0; i < 2; ++i) {
            int mtile = wave;                       // t = wave*2+i -> t>>1 = wave
            int ntile = i;
            int arow = mtile * 16 + (lane & 15);
            int bcol = ntile * 16 + (lane & 15);
            v8f acc = gemm_rowrow<4>(sh_alpha, NTOK, arow, sh_vT, NTOK, bcol, lane, zero_acc());
            int j = bcol;
            if (j < HD) {
                int ch = CDIM + h * HD + j;
                int rbase = mtile * 16 + ((lane >> 4) << 3);
#pragma unroll
                for (int r = 0; r < 8; ++r)
                    sh_cat[(rbase + r) * LDC + ch] = (__bf16)acc[r];
            }
        }
        __syncthreads();
    }

    // ======================= mutual (cross-half) attention ==================
#pragma unroll 1
    for (int h = 0; h < NHEADS; ++h) {
        qkv_stage(sh_xpb, wmutp, h, wave, lane, sh_q, sh_k, sh_vT);
        __syncthreads();

        // beta1 = q2@k1^T (store rows 0..63), beta2 = q1@k2^T (rows 64..127)
#pragma unroll 1
        for (int i = 0; i < 4; ++i) {
            int t = wave * 4 + i;                 // 0..31
            int half2 = __builtin_amdgcn_readfirstlane(t >> 4);
            int tt = t & 15;
            int mtile = __builtin_amdgcn_readfirstlane(tt >> 2);
            int ntile = __builtin_amdgcn_readfirstlane(tt & 3);
            int q_off = half2 ? 0 : 64;
            int k_off = half2 ? 64 : 0;
            int s_off = half2 ? 64 : 0;
            int arow = q_off + mtile * 16 + (lane & 15);
            int bcol = k_off + ntile * 16 + (lane & 15);
            v8f acc = gemm_rowrow<1>(sh_q, HDP, arow, sh_k, HDP, bcol, lane, zero_acc());
            int m = ntile * 16 + (lane & 15);     // store col 0..63
            int rbase = s_off + mtile * 16 + ((lane >> 4) << 3);
#pragma unroll
            for (int r = 0; r < 8; ++r) sh_beta[(rbase + r) * LDBETA + m] = acc[r];
        }
        __syncthreads();

        // softmax over 64 cols, 128 rows
#pragma unroll 1
        for (int i = 0; i < 16; ++i) {
            int row = wave + (i << 3);
            float v0 = sh_beta[row * LDBETA + lane];
            float v1 = sh_beta[row * LDBETA + lane + 32];
            float mx = fmaxf(v0, v1);
            for (int off = 16; off; off >>= 1) mx = fmaxf(mx, __shfl_xor(mx, off));
            float e0 = __expf(v0 - mx), e1 = __expf(v1 - mx);
            float s = e0 + e1;
            for (int off = 16; off; off >>= 1) s += __shfl_xor(s, off);
            float inv = 1.0f / s;
            sh_alpha[row * NTOK + lane]      = (__bf16)(e0 * inv);
            sh_alpha[row * NTOK + lane + 32] = (__bf16)(e1 * inv);
        }
        __syncthreads();

        // x1a = alpha[0:64] @ v1,  x2a = alpha[64:128] @ v2   (K=64, 2 wmma)
#pragma unroll 1
        for (int i = 0; i < 2; ++i) {
            int mtile = wave;
            int ntile = i;
            int vbase = (mtile < 4) ? 0 : 64;
            int arow = mtile * 16 + (lane & 15);
            int bcol = ntile * 16 + (lane & 15);
            v8f acc = gemm_rowrow<2>(sh_alpha, NTOK, arow, sh_vT + vbase, NTOK, bcol,
                                     lane, zero_acc());
            int j = bcol;
            if (j < HD) {
                int ch = h * HD + j;
                int rbase = mtile * 16 + ((lane >> 4) << 3);
#pragma unroll
                for (int r = 0; r < 8; ++r)
                    sh_cat[(rbase + r) * LDC + ch] = (__bf16)acc[r];
            }
        }
        __syncthreads();
    }

    // ---- final projection: (128x288) @ wproj^T + b  -> global f32 ----------
#pragma unroll 1
    for (int i = 0; i < 9; ++i) {
        int mtile = wave;                         // t = wave*9+i -> t/9 = wave
        int ntile = i;
        int arow = mtile * 16 + (lane & 15);
        int bcol = ntile * 16 + (lane & 15);      // out channel (<144)
        v8f acc = gemm_rowrow<9>(sh_cat, LDC, arow, wprojp, 288, bcol, lane, zero_acc());
        float bias = b_proj[bcol];
        int rbase = mtile * 16 + ((lane >> 4) << 3);
#pragma unroll
        for (int r = 0; r < 8; ++r)
            out[((size_t)b * NTOK + rbase + r) * CDIM + bcol] = acc[r] + bias;
    }
}

// ---------------------------------------------------------------------------
extern "C" void kernel_launch(void* const* d_in, const int* in_sizes, int n_in,
                              void* d_out, int out_size, void* d_ws, size_t ws_size,
                              hipStream_t stream) {
    const float* x          = (const float*)d_in[0];
    const float* rpb_table  = (const float*)d_in[1];
    const float* w_qkv_self = (const float*)d_in[2];
    const float* w_qkv_mut  = (const float*)d_in[3];
    const float* w_proj     = (const float*)d_in[4];
    const float* b_proj     = (const float*)d_in[5];
    float*       out        = (float*)d_out;
    __bf16*      ws         = (__bf16*)d_ws;

    int prep_threads = WS_ELEMS;
    int prep_blocks  = (prep_threads + 255) / 256;
    wmsa_prep_kernel<<<prep_blocks, 256, 0, stream>>>(w_qkv_self, w_qkv_mut, w_proj, ws);
    wmsa_main_kernel<<<1024, 256, 0, stream>>>(x, rpb_table, b_proj, ws, out);
}
